// StatelessRNN_6528350289950
// MI455X (gfx1250) — matrix operand order repaired and checked by reading, not compiled, not run on hardware
//
#include <hip/hip_runtime.h>
#include <stdint.h>
#include <stddef.h>

// Problem constants (from reference)
#define B_   128
#define T_   1024
#define F_   64
#define E_   64
#define NU_  104
#define N3_  312          // 3*NU
#define NT_  20           // ceil(312/16) n-tiles (padded to 320 cols)
#define KD_  64           // padded K (=128) as bf16 pairs -> 64 dwords
#define NPK_ 52           // NU_ as bf16 pairs

typedef __attribute__((ext_vector_type(16))) __bf16 v16bf;
typedef __attribute__((ext_vector_type(8)))  float  v8f;

union Frag { v16bf v; uint32_t u[8]; };

__device__ __forceinline__ uint32_t pack_bf2(float lo, float hi) {
  uint32_t a = __float_as_uint(lo), b = __float_as_uint(hi);
  a = (a + 0x7FFFu + ((a >> 16) & 1u)) >> 16;   // round-to-nearest-even bf16
  b = (b + 0x7FFFu + ((b >> 16) & 1u)) >> 16;
  return (a & 0xFFFFu) | (b << 16);
}

__device__ __forceinline__ float sigm_(float x) { return 1.0f / (1.0f + __expf(-x)); }

// ---------------------------------------------------------------------------
// Convert W [K x 312] fp32 (row-major, k-major) into n-major packed-bf16 image
// Wt[320][64] dwords (rows n, 128 bf16 K values, zero padded).
// ---------------------------------------------------------------------------
__global__ void wconv_kernel(const float* __restrict__ W, int K,
                             uint32_t* __restrict__ Wt) {
  int idx = blockIdx.x * blockDim.x + threadIdx.x;   // 320*64 = 20480
  if (idx >= 320 * KD_) return;
  int n = idx / KD_, q = idx % KD_;
  int k0 = 2 * q, k1 = 2 * q + 1;
  float lo = (n < N3_ && k0 < K) ? W[(size_t)k0 * N3_ + n] : 0.0f;
  float hi = (n < N3_ && k1 < K) ? W[(size_t)k1 * N3_ + n] : 0.0f;
  Wt[idx] = pack_bf2(lo, hi);
}

// ---------------------------------------------------------------------------
// One GRU layer. gridDim.x = 8 (16 batch rows each), blockDim.x = 320 (10 waves).
// layer==1: input = concat(x, emb[ids]) (K=128), writes packed-bf16 h sequence.
// layer==2: input = packed h1 sequence (K=104 padded to 128), writes final h fp32.
// ---------------------------------------------------------------------------
__global__ __launch_bounds__(320, 1)
void gru_kernel(int layer,
                const float* __restrict__ x, const int* __restrict__ ids,
                const float* __restrict__ emb,
                const uint32_t* __restrict__ Wt, const uint32_t* __restrict__ Ut,
                const float* __restrict__ bias,             // [2][312]
                const uint32_t* __restrict__ h1pk_in,       // layer 2 input
                uint32_t* __restrict__ h1pk_out,            // layer 1 output
                float* __restrict__ h_final) {              // layer 2 output
  __shared__ uint32_t xcat[16][KD_];     // per-step input, packed bf16
  __shared__ uint32_t hbf[16][KD_];      // h state, packed bf16 (zero padded)
  __shared__ float    xw[16][NT_ * 16];  // x @ W + b_in      (320 cols)
  __shared__ float    mbuf[16][NT_ * 16];// h @ U + b_rec     (320 cols)
  __shared__ float    hst[16][NU_];      // h state fp32

  const int tid  = threadIdx.x;
  const int lane = tid & 31;
  const int wv   = tid >> 5;             // 0..9
  const int hf   = lane >> 4;            // lane half
  const int mr   = lane & 15;
  const int b0   = blockIdx.x * 16;

  // A-fragment (16x32 bf16) pair indices per ISA 7.12.2:
  // v<4 : K = hf?8:0  + 2v     -> pair hf*4 + v
  // v>=4: K = hf?24:16 + 2(v-4)-> pair 8 + hf*4 + (v-4)
  int pidx[8];
#pragma unroll
  for (int v = 0; v < 8; ++v)
    pidx[v] = (v < 4) ? (hf * 4 + v) : (8 + hf * 4 + (v - 4));

  // init state
  for (int i = tid; i < 16 * KD_; i += 320) ((uint32_t*)hbf)[i] = 0u;
  for (int i = tid; i < 16 * NU_; i += 320) ((float*)hst)[i] = 0.0f;
  __syncthreads();

  for (int t = 0; t < T_; ++t) {
    // ---- phase 1: build packed-bf16 input tile [16][128] -------------------
    if (layer == 1) {
      for (int i = tid; i < 16 * KD_; i += 320) {
        int m = i >> 6, q = i & 63;
        int b = b0 + m;
        float lo, hi;
        if (q < 32) {                       // x part (F=64 -> 32 pairs)
          const float* xr = x + ((size_t)b * T_ + t) * F_;
          lo = xr[2 * q]; hi = xr[2 * q + 1];
        } else {                            // embedding part (E=64 -> 32 pairs)
          int id = ids[(size_t)b * T_ + t];
          const float* er = emb + (size_t)id * E_;
          int qq = q - 32;
          lo = er[2 * qq]; hi = er[2 * qq + 1];
        }
        xcat[m][q] = pack_bf2(lo, hi);
      }
    } else {
      for (int i = tid; i < 16 * KD_; i += 320) {
        int m = i >> 6, q = i & 63;
        int b = b0 + m;
        xcat[m][q] = (q < NPK_) ? h1pk_in[((size_t)b * T_ + t) * NPK_ + q] : 0u;
      }
    }
    __syncthreads();

    // ---- phase 2: two WMMA GEMMs: xw = in@W + b0 ; mbuf = h@U + b1 ---------
    {
      Frag ax[4], ah[4];
      const uint32_t* xrow = &xcat[mr][0];
      const uint32_t* hrow = &hbf[mr][0];
#pragma unroll
      for (int ks = 0; ks < 4; ++ks)
#pragma unroll
        for (int v = 0; v < 8; ++v) {
          ax[ks].u[v] = xrow[ks * 16 + pidx[v]];
          ah[ks].u[v] = hrow[ks * 16 + pidx[v]];
        }

#pragma unroll
      for (int s = 0; s < 2; ++s) {
        int nt = wv * 2 + s;                     // 10 waves * 2 = 20 tiles
        const uint32_t* wb = Wt + (size_t)(nt * 16 + mr) * KD_;
        const uint32_t* ub = Ut + (size_t)(nt * 16 + mr) * KD_;
        v8f accx = {};
        v8f acch = {};
#pragma unroll
        for (int ks = 0; ks < 4; ++ks) {
          Frag bw, bu;                            // B 32x16: pair = pidx[v] (same as A)
#pragma unroll
          for (int v = 0; v < 8; ++v) {
            bw.u[v] = wb[ks * 16 + pidx[v]];
            bu.u[v] = ub[ks * 16 + pidx[v]];
          }
          accx = __builtin_amdgcn_wmma_f32_16x16x32_bf16(
              false, ax[ks].v, false, bw.v, (short)0, accx, false, false);
          acch = __builtin_amdgcn_wmma_f32_16x16x32_bf16(
              false, ah[ks].v, false, bu.v, (short)0, acch, false, false);
        }
        int ncol = nt * 16 + mr;                  // C/D: lane = N, VGPR = M
        if (ncol < N3_) {
          float bi = bias[ncol];                  // input bias row
          float br = bias[N3_ + ncol];            // recurrent bias row
#pragma unroll
          for (int v = 0; v < 8; ++v) {
            xw[hf * 8 + v][ncol]   = accx[v] + bi;
            mbuf[hf * 8 + v][ncol] = acch[v] + br;
          }
        }
      }
    }
    __syncthreads();

    // ---- phase 3: gates (fp32) --------------------------------------------
    for (int i = tid; i < 16 * NU_; i += 320) {
      int m = i / NU_, j = i % NU_;
      float z  = sigm_(xw[m][j]           + mbuf[m][j]);
      float r  = sigm_(xw[m][NU_ + j]     + mbuf[m][NU_ + j]);
      float hh = tanhf(xw[m][2 * NU_ + j] + r * mbuf[m][2 * NU_ + j]);
      float hn = z * hst[m][j] + (1.0f - z) * hh;
      hst[m][j] = hn;
    }
    __syncthreads();

    // ---- phase 4: repack h -> bf16 (and emit sequence for layer 1) --------
    for (int i = tid; i < 16 * NPK_; i += 320) {
      int m = i / NPK_, q = i % NPK_;
      uint32_t pk = pack_bf2(hst[m][2 * q], hst[m][2 * q + 1]);
      hbf[m][q] = pk;
      if (layer == 1)
        h1pk_out[(((size_t)(b0 + m)) * T_ + t) * NPK_ + q] = pk;
    }
    // next iteration's phase-1 barrier orders these writes before the GEMMs
  }

  if (layer == 2) {
    __syncthreads();
    for (int i = tid; i < 16 * NU_; i += 320) {
      int m = i / NU_, j = i % NU_;
      h_final[(size_t)(b0 + m) * NU_ + j] = hst[m][j];
    }
  }
}

// ---------------------------------------------------------------------------
// Head: out[b] = sigmoid(h2[b,:] . Wo + bo)
// ---------------------------------------------------------------------------
__global__ void head_kernel(const float* __restrict__ h2,
                            const float* __restrict__ Wo,
                            const float* __restrict__ bo,
                            float* __restrict__ out) {
  int b = blockIdx.x * blockDim.x + threadIdx.x;
  if (b >= B_) return;
  float s = bo[0];
#pragma unroll 8
  for (int j = 0; j < NU_; ++j) s += h2[(size_t)b * NU_ + j] * Wo[j];
  out[b] = sigm_(s);
}

extern "C" void kernel_launch(void* const* d_in, const int* in_sizes, int n_in,
                              void* d_out, int out_size, void* d_ws, size_t ws_size,
                              hipStream_t stream) {
  (void)in_sizes; (void)n_in; (void)out_size; (void)ws_size;
  const float* x   = (const float*)d_in[0];
  const int*   ids = (const int*)d_in[1];
  const float* emb = (const float*)d_in[2];
  const float* W1  = (const float*)d_in[3];
  const float* U1  = (const float*)d_in[4];
  const float* b1  = (const float*)d_in[5];
  const float* W2  = (const float*)d_in[6];
  const float* U2  = (const float*)d_in[7];
  const float* b2  = (const float*)d_in[8];
  const float* Wo  = (const float*)d_in[9];
  const float* bo  = (const float*)d_in[10];

  uint8_t* ws = (uint8_t*)d_ws;
  const size_t WSZ = (size_t)320 * KD_ * 4;            // 81920 B per weight image
  uint32_t* W1t  = (uint32_t*)(ws + 0 * WSZ);
  uint32_t* U1t  = (uint32_t*)(ws + 1 * WSZ);
  uint32_t* W2t  = (uint32_t*)(ws + 2 * WSZ);
  uint32_t* U2t  = (uint32_t*)(ws + 3 * WSZ);
  float*    h2   = (float*)(ws + 4 * WSZ);             // 128*104*4 = 53248 B
  uint32_t* h1pk = (uint32_t*)(ws + 4 * WSZ + 65536);  // 128*1024*52*4 = 27.3 MB

  // One-time (per call) weight conversion to n-major packed bf16.
  wconv_kernel<<<80, 256, 0, stream>>>(W1, F_ + E_, W1t);
  wconv_kernel<<<80, 256, 0, stream>>>(U1, NU_,     U1t);
  wconv_kernel<<<80, 256, 0, stream>>>(W2, NU_,     W2t);
  wconv_kernel<<<80, 256, 0, stream>>>(U2, NU_,     U2t);

  // Layer 1: emits packed-bf16 hidden sequence.
  gru_kernel<<<B_ / 16, 320, 0, stream>>>(1, x, ids, emb, W1t, U1t, b1,
                                          nullptr, h1pk, nullptr);
  // Layer 2: consumes the sequence, emits final fp32 state.
  gru_kernel<<<B_ / 16, 320, 0, stream>>>(2, x, ids, emb, W2t, U2t, b2,
                                          h1pk, nullptr, h2);
  // Head.
  head_kernel<<<1, 128, 0, stream>>>(h2, Wo, bo, (float*)d_out);
}
